// DecoderRNN_10788957848190
// MI455X (gfx1250) — matrix operand order repaired and checked by reading, not compiled, not run on hardware
//
#include <hip/hip_runtime.h>
#include <hip/hip_bf16.h>
#include <math.h>

// Sizes from the reference
#define V_SZ 4096
#define E_SZ 512
#define H_SZ 1024
#define B_SZ 16
#define T_SZ 48

typedef __attribute__((ext_vector_type(16))) __bf16 v16bf;
typedef __attribute__((ext_vector_type(8)))  __bf16 v8bf;
typedef __attribute__((ext_vector_type(8)))  float  v8f;

__device__ __forceinline__ float sigmoidf_(float x) { return 1.0f / (1.0f + expf(-x)); }

// ---------------------------------------------------------------------------
// Weight-norm row scales: scales[r] = g[r] / ||v_row||  (rows 0..4095: lin_v,
// rows 4096..8191: lin1_v)
// ---------------------------------------------------------------------------
__global__ void row_scales_kernel(const float* __restrict__ lin_v,
                                  const float* __restrict__ lin_g,
                                  const float* __restrict__ lin1_v,
                                  const float* __restrict__ lin1_g,
                                  float* __restrict__ scales) {
    __shared__ float sred[256];
    const int r = blockIdx.x;
    const int tid = threadIdx.x;
    const float* src;
    const float* g;
    int K, gi;
    if (r < V_SZ) { src = lin_v  + (size_t)r * H_SZ;          K = H_SZ; g = lin_g;  gi = r; }
    else          { src = lin1_v + (size_t)(r - V_SZ) * V_SZ; K = V_SZ; g = lin1_g; gi = r - V_SZ; }
    float sq = 0.f;
    for (int k = tid; k < K; k += 256) { float v = src[k]; sq += v * v; }
    sred[tid] = sq; __syncthreads();
    for (int off = 128; off > 0; off >>= 1) {
        if (tid < off) sred[tid] += sred[tid + off];
        __syncthreads();
    }
    if (tid == 0) scales[r] = g[gi] / sqrtf(sred[0]);
}

// ---------------------------------------------------------------------------
// Pack an (N=4096, K) fp32 row-major weight matrix into bf16 WMMA B-fragment
// tiles. Tile (nt,kt) covers cols n=nt*16..+15, rows k=kt*32..+31 of B=W^T.
// Lane layout (B, 16-bit, 16x16x32): lanes 0-15 -> col n=lane, K={0..7,16..23};
// lanes 16-31 -> col n=lane-16, K={8..15,24..31}; 16 bf16 contiguous per lane.
// ---------------------------------------------------------------------------
__global__ void pack_w_kernel(const float* __restrict__ src, int K,
                              const float* __restrict__ scale,
                              __bf16* __restrict__ dst) {
    const int tid  = blockIdx.x * 256 + threadIdx.x;
    const int lane = tid & 31;
    const int tile = tid >> 5;
    const int KT = K >> 5;
    const int nt = tile / KT;
    const int kt = tile - nt * KT;
    const int n = nt * 16 + (lane & 15);
    const int kbase = kt * 32 + ((lane >= 16) ? 8 : 0);
    const float sc = scale ? scale[n] : 1.0f;
    const float* s = src + (size_t)n * K + kbase;
    v16bf d;
#pragma unroll
    for (int j = 0; j < 16; ++j) d[j] = (__bf16)(s[j + (j & 8)] * sc);
    *(v16bf*)(dst + (size_t)tile * 512 + lane * 16) = d;
}

// ---------------------------------------------------------------------------
// Init: c=0, h_bf16=0, x_bf16 = bf16(features)
// ---------------------------------------------------------------------------
__global__ void init_kernel(const float* __restrict__ features,
                            float* __restrict__ c,
                            __bf16* __restrict__ hbf,
                            __bf16* __restrict__ xbf) {
    const int idx = blockIdx.x * 256 + threadIdx.x;  // 16384 threads
    c[idx] = 0.f;
    hbf[idx] = (__bf16)0.f;
    if (idx < B_SZ * E_SZ) xbf[idx] = (__bf16)features[idx];
}

// ---------------------------------------------------------------------------
// A-fragment loader: X is (16 x K) bf16 row-major. Documented dense-A layout:
// lanes 0-15: row m=lane, K halves {0..7,16..23}; lanes 16-31: row m=lane-16,
// K halves {8..15,24..31}. Two aligned 16B loads.
// ---------------------------------------------------------------------------
__device__ __forceinline__ v16bf load_a_frag(const __bf16* __restrict__ X,
                                             int K, int kt, int lane) {
    const int brow = lane & 15;
    const int koff = (lane < 16) ? 0 : 8;
    const __bf16* ap = X + (size_t)brow * K + kt * 32 + koff;
    v8bf a0 = *(const v8bf*)(ap);
    v8bf a1 = *(const v8bf*)(ap + 16);
    v16bf a;
#pragma unroll
    for (int j = 0; j < 8; ++j) { a[j] = a0[j]; a[j + 8] = a1[j]; }
    return a;
}

__device__ __forceinline__ v16bf load_b_frag(const __bf16* __restrict__ Wt,
                                             int KT, int nt, int kt, int lane) {
    return *(const v16bf*)(Wt + ((size_t)nt * KT + kt) * 512 + lane * 16);
}

// ---------------------------------------------------------------------------
// Reduce 4 wave-level accumulators through LDS and store the 16x16 tile.
// C/D layout: VGPR v, lanes 0-15 -> (M=v, N=lane); lanes 16-31 -> (M=v+8).
// ---------------------------------------------------------------------------
__device__ __forceinline__ void reduce_store(float* __restrict__ red,
                                             v8f acc, int wave, int lane,
                                             int nt, float* __restrict__ out, int N) {
#pragma unroll
    for (int v = 0; v < 8; ++v) red[wave * 256 + v * 32 + lane] = acc[v];
    __syncthreads();
    if (wave == 0) {
#pragma unroll
        for (int v = 0; v < 8; ++v) {
            float s = red[v * 32 + lane] + red[256 + v * 32 + lane] +
                      red[512 + v * 32 + lane] + red[768 + v * 32 + lane];
            const int row = v + ((lane >= 16) ? 8 : 0);
            const int col = nt * 16 + (lane & 15);
            out[(size_t)row * N + col] = s;
        }
    }
}

// ---------------------------------------------------------------------------
// Generic GEMM: out(16 x 4096) = X(16 x K) @ Wtiles.  One WG (4 waves) per
// 16-col tile, compile-time K split across waves (uniform trip count -> no
// EXEC masking), dual accumulators to break the WMMA D->C chain.
// ---------------------------------------------------------------------------
template <int K>
__global__ void gemm16_kernel(const __bf16* __restrict__ X,
                              const __bf16* __restrict__ Wt,
                              float* __restrict__ out) {
    constexpr int KT = K / 32;          // k-tiles total
    constexpr int PAIRS = KT / 8;       // per-wave tile pairs
    __shared__ float red[4 * 256];
    const int lane = threadIdx.x & 31;
    const int wave = threadIdx.x >> 5;
    const int nt = blockIdx.x;
    v8f acc0 = {}, acc1 = {};
#pragma unroll 4
    for (int i = 0; i < PAIRS; ++i) {
        const int kt0 = wave + 8 * i;
        const int kt1 = kt0 + 4;
        __builtin_prefetch(Wt + ((size_t)nt * KT + kt0 + 8) * 512 + lane * 16, 0, 3);
        v16bf a0 = load_a_frag(X, K, kt0, lane);
        v16bf b0 = load_b_frag(Wt, KT, nt, kt0, lane);
        v16bf a1 = load_a_frag(X, K, kt1, lane);
        v16bf b1 = load_b_frag(Wt, KT, nt, kt1, lane);
        acc0 = __builtin_amdgcn_wmma_f32_16x16x32_bf16(false, a0, false, b0,
                                                       (short)0, acc0, false, false);
        acc1 = __builtin_amdgcn_wmma_f32_16x16x32_bf16(false, a1, false, b1,
                                                       (short)0, acc1, false, false);
    }
#pragma unroll
    for (int v = 0; v < 8; ++v) acc0[v] += acc1[v];
    reduce_store(red, acc0, wave, lane, nt, out, V_SZ);
}

// ---------------------------------------------------------------------------
// Fused gates GEMM: out(16 x 4096) = x(16x512)@Wih^T + h(16x1024)@Whh^T.
// Compile-time loops: Wih 4 tiles/wave, Whh 8 tiles/wave, dual accumulators.
// ---------------------------------------------------------------------------
__global__ void gemm_gates_kernel(const __bf16* __restrict__ Xe,
                                  const __bf16* __restrict__ Hh,
                                  const __bf16* __restrict__ WihP,
                                  const __bf16* __restrict__ WhhP,
                                  float* __restrict__ out) {
    __shared__ float red[4 * 256];
    const int lane = threadIdx.x & 31;
    const int wave = threadIdx.x >> 5;
    const int nt = blockIdx.x;
    v8f acc0 = {}, acc1 = {};
#pragma unroll
    for (int i = 0; i < 2; ++i) {       // Wih: KT=16
        const int kt0 = wave + 8 * i;
        const int kt1 = kt0 + 4;
        v16bf a0 = load_a_frag(Xe, E_SZ, kt0, lane);
        v16bf b0 = load_b_frag(WihP, 16, nt, kt0, lane);
        v16bf a1 = load_a_frag(Xe, E_SZ, kt1, lane);
        v16bf b1 = load_b_frag(WihP, 16, nt, kt1, lane);
        acc0 = __builtin_amdgcn_wmma_f32_16x16x32_bf16(false, a0, false, b0,
                                                       (short)0, acc0, false, false);
        acc1 = __builtin_amdgcn_wmma_f32_16x16x32_bf16(false, a1, false, b1,
                                                       (short)0, acc1, false, false);
    }
#pragma unroll
    for (int i = 0; i < 4; ++i) {       // Whh: KT=32
        const int kt0 = wave + 8 * i;
        const int kt1 = kt0 + 4;
        v16bf a0 = load_a_frag(Hh, H_SZ, kt0, lane);
        v16bf b0 = load_b_frag(WhhP, 32, nt, kt0, lane);
        v16bf a1 = load_a_frag(Hh, H_SZ, kt1, lane);
        v16bf b1 = load_b_frag(WhhP, 32, nt, kt1, lane);
        acc0 = __builtin_amdgcn_wmma_f32_16x16x32_bf16(false, a0, false, b0,
                                                       (short)0, acc0, false, false);
        acc1 = __builtin_amdgcn_wmma_f32_16x16x32_bf16(false, a1, false, b1,
                                                       (short)0, acc1, false, false);
    }
#pragma unroll
    for (int v = 0; v < 8; ++v) acc0[v] += acc1[v];
    reduce_store(red, acc0, wave, lane, nt, out, 4 * H_SZ);
}

// ---------------------------------------------------------------------------
// LSTM elementwise: c' = sig(f)*c + sig(i)*tanh(g); h' = sig(o)*tanh(c')
// ---------------------------------------------------------------------------
__global__ void lstm_cell_kernel(const float* __restrict__ gates,
                                 const float* __restrict__ b_ih,
                                 const float* __restrict__ b_hh,
                                 float* __restrict__ c,
                                 __bf16* __restrict__ hbf,
                                 float* __restrict__ hs, int t) {
    const int idx = blockIdx.x * 256 + threadIdx.x;  // 16384
    const int b = idx >> 10, j = idx & (H_SZ - 1);
    const float* gr = gates + (size_t)b * 4 * H_SZ;
    const float ig = gr[j]            + b_ih[j]            + b_hh[j];
    const float fg = gr[H_SZ + j]     + b_ih[H_SZ + j]     + b_hh[H_SZ + j];
    const float gg = gr[2 * H_SZ + j] + b_ih[2 * H_SZ + j] + b_hh[2 * H_SZ + j];
    const float og = gr[3 * H_SZ + j] + b_ih[3 * H_SZ + j] + b_hh[3 * H_SZ + j];
    const float cn = sigmoidf_(fg) * c[idx] + sigmoidf_(ig) * tanhf(gg);
    const float hn = sigmoidf_(og) * tanhf(cn);
    c[idx] = cn;
    hbf[idx] = (__bf16)hn;
    hs[((size_t)b * T_SZ + t) * H_SZ + j] = hn;
}

// ---------------------------------------------------------------------------
// LayerNorm + leaky_relu(0.3) over V=4096 per row -> z (bf16)
// ---------------------------------------------------------------------------
__global__ void ln_lrelu_kernel(const float* __restrict__ acc,
                                const float* __restrict__ lin_b,
                                const float* __restrict__ g,
                                const float* __restrict__ bb,
                                __bf16* __restrict__ z) {
    __shared__ float s1[256], s2[256];
    const int b = blockIdx.x, tid = threadIdx.x;
    float vals[16];
    float sum = 0.f, sq = 0.f;
#pragma unroll
    for (int i = 0; i < 16; ++i) {
        const int n = tid + i * 256;
        const float v = acc[(size_t)b * V_SZ + n] + lin_b[n];
        vals[i] = v; sum += v; sq += v * v;
    }
    s1[tid] = sum; s2[tid] = sq; __syncthreads();
    for (int off = 128; off > 0; off >>= 1) {
        if (tid < off) { s1[tid] += s1[tid + off]; s2[tid] += s2[tid + off]; }
        __syncthreads();
    }
    const float mu = s1[0] * (1.0f / V_SZ);
    const float var = s2[0] * (1.0f / V_SZ) - mu * mu;
    const float inv = rsqrtf(var + 1e-5f);
#pragma unroll
    for (int i = 0; i < 16; ++i) {
        const int n = tid + i * 256;
        float y = (vals[i] - mu) * inv * g[n] + bb[n];
        y = (y > 0.f) ? y : 0.3f * y;
        z[(size_t)b * V_SZ + n] = (__bf16)y;
    }
}

// ---------------------------------------------------------------------------
// LayerNorm + softmax + argmax + embedding gather (feeds next step)
// ---------------------------------------------------------------------------
__global__ void ln_softmax_emb_kernel(const float* __restrict__ acc,
                                      const float* __restrict__ lin1_b,
                                      const float* __restrict__ g,
                                      const float* __restrict__ bb,
                                      const float* __restrict__ embed,
                                      float* __restrict__ Q,
                                      float* __restrict__ ge,
                                      float* __restrict__ preds,
                                      __bf16* __restrict__ xnext, int t) {
    __shared__ float s1[256], s2[256];
    __shared__ int si[256];
    const int b = blockIdx.x, tid = threadIdx.x;
    float vals[16];
    float sum = 0.f, sq = 0.f;
#pragma unroll
    for (int i = 0; i < 16; ++i) {
        const int n = tid + i * 256;
        const float v = acc[(size_t)b * V_SZ + n] + lin1_b[n];
        vals[i] = v; sum += v; sq += v * v;
    }
    s1[tid] = sum; s2[tid] = sq; __syncthreads();
    for (int off = 128; off > 0; off >>= 1) {
        if (tid < off) { s1[tid] += s1[tid + off]; s2[tid] += s2[tid + off]; }
        __syncthreads();
    }
    const float mu = s1[0] * (1.0f / V_SZ);
    const float var = s2[0] * (1.0f / V_SZ) - mu * mu;
    const float inv = rsqrtf(var + 1e-5f);
    float bmax = -3.4e38f; int bidx = 0;
#pragma unroll
    for (int i = 0; i < 16; ++i) {
        const int n = tid + i * 256;
        const float y = (vals[i] - mu) * inv * g[n] + bb[n];
        vals[i] = y;
        if (y > bmax) { bmax = y; bidx = n; }
    }
    __syncthreads();
    s1[tid] = bmax; si[tid] = bidx; __syncthreads();
    for (int off = 128; off > 0; off >>= 1) {
        if (tid < off) {
            const float ov = s1[tid + off]; const int oi = si[tid + off];
            if (ov > s1[tid] || (ov == s1[tid] && oi < si[tid])) { s1[tid] = ov; si[tid] = oi; }
        }
        __syncthreads();
    }
    const float rmax = s1[0];
    const int pred = si[0];
    __syncthreads();
    float es = 0.f;
#pragma unroll
    for (int i = 0; i < 16; ++i) { const float e = expf(vals[i] - rmax); vals[i] = e; es += e; }
    s2[tid] = es; __syncthreads();
    for (int off = 128; off > 0; off >>= 1) {
        if (tid < off) s2[tid] += s2[tid + off];
        __syncthreads();
    }
    const float dinv = 1.0f / s2[0];
    float* Qrow = Q + ((size_t)b * T_SZ + t) * V_SZ;
#pragma unroll
    for (int i = 0; i < 16; ++i) Qrow[tid + i * 256] = vals[i] * dinv;
    // embedding gather -> ge output and next-step x (bf16)
    const float* erow = embed + (size_t)pred * E_SZ;
    float* gerow = ge + ((size_t)b * T_SZ + t) * E_SZ;
    for (int e = tid; e < E_SZ; e += 256) {
        const float ev = erow[e];
        gerow[e] = ev;
        xnext[(size_t)b * E_SZ + e] = (__bf16)ev;
    }
    if (tid == 0) preds[(size_t)b * T_SZ + t] = (float)pred;
}

// ---------------------------------------------------------------------------
extern "C" void kernel_launch(void* const* d_in, const int* in_sizes, int n_in,
                              void* d_out, int out_size, void* d_ws, size_t ws_size,
                              hipStream_t stream) {
    const float* features = (const float*)d_in[0];
    // d_in[1] captions: unused by the greedy decode path (only defines T=48)
    const float* embed    = (const float*)d_in[2];
    const float* W_ih     = (const float*)d_in[3];
    const float* W_hh     = (const float*)d_in[4];
    const float* b_ih     = (const float*)d_in[5];
    const float* b_hh     = (const float*)d_in[6];
    const float* lin_v    = (const float*)d_in[7];
    const float* lin_g    = (const float*)d_in[8];
    const float* lin_b    = (const float*)d_in[9];
    const float* lin1_v   = (const float*)d_in[10];
    const float* lin1_g   = (const float*)d_in[11];
    const float* lin1_b   = (const float*)d_in[12];
    const float* ln_g     = (const float*)d_in[13];
    const float* ln_b     = (const float*)d_in[14];
    const float* ln1_g    = (const float*)d_in[15];
    const float* ln1_b    = (const float*)d_in[16];

    // Workspace layout (all sizes power-of-two aligned). ~53 MB total:
    // all bf16 weights fit in the 192 MB L2 so the 48-step loop streams
    // from L2, not HBM.
    char* ws = (char*)d_ws;
    __bf16* WihP  = (__bf16*)ws;                 ws += (size_t)(4 * H_SZ) * E_SZ * 2;  // 4 MB
    __bf16* WhhP  = (__bf16*)ws;                 ws += (size_t)(4 * H_SZ) * H_SZ * 2;  // 8 MB
    __bf16* WP    = (__bf16*)ws;                 ws += (size_t)V_SZ * H_SZ * 2;        // 8 MB
    __bf16* W1P   = (__bf16*)ws;                 ws += (size_t)V_SZ * V_SZ * 2;        // 32 MB
    float*  scales= (float*)ws;                  ws += 2 * V_SZ * 4;
    float*  accb  = (float*)ws;                  ws += (size_t)B_SZ * V_SZ * 4;
    float*  cbuf  = (float*)ws;                  ws += (size_t)B_SZ * H_SZ * 4;
    __bf16* hbf   = (__bf16*)ws;                 ws += (size_t)B_SZ * H_SZ * 2;
    __bf16* xbf   = (__bf16*)ws;                 ws += (size_t)B_SZ * E_SZ * 2;
    __bf16* zbf   = (__bf16*)ws;                 ws += (size_t)B_SZ * V_SZ * 2;

    float* out = (float*)d_out;
    float* Q   = out;                                        // (B,T,V)
    float* hs  = out + (size_t)B_SZ * T_SZ * V_SZ;           // (B,T,H)
    float* ge  = hs  + (size_t)B_SZ * T_SZ * H_SZ;           // (B,T,E)
    float* pr  = ge  + (size_t)B_SZ * T_SZ * E_SZ;           // (B,T)

    // --- Per-launch weight prep (deterministic, recomputed each call) ---
    row_scales_kernel<<<2 * V_SZ, 256, 0, stream>>>(lin_v, lin_g, lin1_v, lin1_g, scales);
    pack_w_kernel<<< 512, 256, 0, stream>>>(W_ih,  E_SZ, nullptr,        WihP);  // 4096 tiles
    pack_w_kernel<<<1024, 256, 0, stream>>>(W_hh,  H_SZ, nullptr,        WhhP);  // 8192 tiles
    pack_w_kernel<<<1024, 256, 0, stream>>>(lin_v, H_SZ, scales,         WP);
    pack_w_kernel<<<4096, 256, 0, stream>>>(lin1_v,V_SZ, scales + V_SZ,  W1P);   // 32768 tiles
    init_kernel<<<64, 256, 0, stream>>>(features, cbuf, hbf, xbf);

    // --- Sequential decode: 48 steps, kernel boundaries are the syncs ---
    for (int t = 0; t < T_SZ; ++t) {
        gemm_gates_kernel<<<256, 128, 0, stream>>>(xbf, hbf, WihP, WhhP, accb);
        lstm_cell_kernel<<<64, 256, 0, stream>>>(accb, b_ih, b_hh, cbuf, hbf, hs, t);
        gemm16_kernel<H_SZ><<<256, 128, 0, stream>>>(hbf, WP, accb);
        ln_lrelu_kernel<<<B_SZ, 256, 0, stream>>>(accb, lin_b, ln_g, ln_b, zbf);
        gemm16_kernel<V_SZ><<<256, 128, 0, stream>>>(zbf, W1P, accb);
        ln_softmax_emb_kernel<<<B_SZ, 256, 0, stream>>>(accb, lin1_b, ln1_g, ln1_b,
                                                        embed, Q, ge, pr, xbf, t);
    }
}
